// sccnn_conv_no_b2_670014898738
// MI455X (gfx1250) — matrix operand order repaired and checked by reading, not compiled, not run on hardware
//
#include <hip/hip_runtime.h>
#include <stdint.h>

// ---------------------------------------------------------------------------
// SCCNN conv forward for MI455X (gfx1250, wave32, WMMA + TDM).
// Heavy GEMMs run as bf16 WMMA / f32 accumulate. The streamed (Laplacian)
// operand is DMA'd into LDS by the Tensor Data Mover, double buffered, and
// converted f32->bf16 per-lane during fragment assembly. Both Chebyshev
// chains per order share one 128-wide RHS so each Laplacian is streamed from
// HBM only twice (512 MB total for L1l -> ~22us roofline @ 23.3 TB/s).
// ---------------------------------------------------------------------------

typedef __attribute__((ext_vector_type(16))) __bf16 bf16x16;
typedef __attribute__((ext_vector_type(8)))  float  f32x8;
typedef __attribute__((ext_vector_type(4)))  unsigned int u32x4;
typedef __attribute__((ext_vector_type(8)))  int    i32x8;
typedef __attribute__((ext_vector_type(4)))  int    i32x4;

#if defined(__has_builtin)
#if __has_builtin(__builtin_amdgcn_tensor_load_to_lds) && \
    __has_builtin(__builtin_amdgcn_s_wait_tensorcnt)
#define SCCNN_HAVE_TDM 1
#endif
#endif
#ifndef SCCNN_HAVE_TDM
#define SCCNN_HAVE_TDM 0
#endif

union AFragU { uint4 u4[2]; bf16x16 v; };
union BFragU { uint4 u4[2]; bf16x16 v; };

#define GEMM_MBLK 64
#define GEMM_KB   32
#define APITCH    36   // dwords per LDS row of f32 A tile (32 data + 4 pad)

// --------------------------- TDM staging -----------------------------------
// Stage a [64 x 32] f32 tile, row stride lda elements, into LDS at lds_dst
// with APITCH-dword row pitch (TDM pad: 4 dwords after every 32 dwords).
__device__ __forceinline__ void stage_A(const float* gsrc, float* lds_dst,
                                        int lda, int tid, int wave) {
#if SCCNN_HAVE_TDM
  if (wave == 0) {
    uint64_t ga = (uint64_t)(size_t)gsrc;
    u32x4 g0;
    g0[0] = 1u;                                   // count=1 (valid user D#)
    g0[1] = (uint32_t)(size_t)lds_dst;            // LDS byte address
    g0[2] = (uint32_t)ga;                         // global addr [31:0]
    g0[3] = (uint32_t)((ga >> 32) & 0x01FFFFFFu)  // global addr [56:32]
          | (2u << 30);                           // type = 2 ("image")
    i32x8 g1;
    // data_size=4B(code2), pad_enable, pad_interval=32dw(code4), pad=4dw(code3)
    g1[0] = (int)((2u << 16) | (1u << 20) | (4u << 22) | (3u << 25));
    uint32_t dim0 = (uint32_t)lda;                // tensor dim0 (row length)
    uint32_t dim1 = 1u << 20;                     // plenty of rows (no OOB)
    uint64_t str0 = (uint64_t)(uint32_t)lda;      // dim0 stride (elements)
    g1[1] = (int)((dim0 & 0xffffu) << 16);        // [63:48]=dim0 lo16
    g1[2] = (int)((dim0 >> 16) | ((dim1 & 0xffffu) << 16));
    g1[3] = (int)((dim1 >> 16) | (32u << 16));    // [127:112]=tile_dim0=32
    g1[4] = (int)64;                              // tile_dim1=64, tile_dim2=0
    g1[5] = (int)(uint32_t)(str0 & 0xffffffffu);
    g1[6] = (int)(uint32_t)((str0 >> 32) & 0xffffu);
    g1[7] = 0;
    i32x4 z4; z4[0] = 0; z4[1] = 0; z4[2] = 0; z4[3] = 0;
    i32x8 z8;
    #pragma unroll
    for (int j = 0; j < 8; ++j) z8[j] = 0;
    __builtin_amdgcn_tensor_load_to_lds(g0, g1, z4, z4, z8, 0);
  }
#else
  #pragma unroll
  for (int r = 0; r < 4; ++r) {
    int idx = tid + r * 128;
    int row = idx >> 3;
    int c4  = (idx & 7) << 2;
    *reinterpret_cast<float4*>(&lds_dst[row * APITCH + c4]) =
        *reinterpret_cast<const float4*>(gsrc + (size_t)row * lda + c4);
  }
  (void)wave;
#endif
}

__device__ __forceinline__ void stage_A_fence(int wave) {
#if SCCNN_HAVE_TDM
  if (wave == 0) __builtin_amdgcn_s_wait_tensorcnt(0);
#else
  (void)wave;
#endif
}

// ------------------- WMMA GEMM, A streamed via TDM -------------------------
// C[M x N](f32) = A[M x K](f32 row-major) * B[K x N](bf16)
// Epilogue: optional rowscale; optional (2*acc - base); optional bf16 mirror.
template<int NT>                                   // N = NT*16 (64 or 128)
__global__ __launch_bounds__(128)
void gemm_tdm_wmma(const float* __restrict__ A, int lda,
                   const __bf16* __restrict__ B, int ldb, int K,
                   float* __restrict__ C, int ldc,
                   __bf16* __restrict__ Cbf, int ldcb,
                   const float* __restrict__ rowscale,
                   const float* __restrict__ base, int ldbase)
{
  constexpr int N      = NT * 16;
  constexpr int BSLOT  = 12;                  // dwords per (tile,lane) slot
  constexpr int NTASKS = (16 * (N / 4)) / 128;

  __shared__ __align__(16) float    Afs[2][GEMM_MBLK * APITCH]; // f32 tiles
  __shared__ __align__(16) uint32_t Bls[2][NT * 32 * BSLOT];    // bf16 frags

  const int tid  = threadIdx.x;
  const int wave = tid >> 5;
  const int lane = tid & 31;
  const int m0   = blockIdx.x * GEMM_MBLK;

  f32x8 acc[NT];
  #pragma unroll
  for (int t = 0; t < NT; ++t) {
    #pragma unroll
    for (int j = 0; j < 8; ++j) acc[t][j] = 0.0f;
  }

  // B staging task geometry: task = (kp, n0) pair; kp in [0,16), n0 step 4.
  auto b_load = [&](int k0, int j, uint2& q0, uint2& q1) {
    int t4 = tid + j * 128;
    int kp = t4 / (N / 4);
    int n0 = (t4 % (N / 4)) << 2;
    const __bf16* b0 = B + (size_t)(k0 + 2 * kp) * ldb + n0;
    q0 = *reinterpret_cast<const uint2*>(b0);
    q1 = *reinterpret_cast<const uint2*>(b0 + ldb);
  };
  auto b_store = [&](int buf, int j, uint2 q0, uint2 q1) {
    int t4 = tid + j * 128;
    int kp = t4 / (N / 4);
    int n0 = (t4 % (N / 4)) << 2;
    int tt    = n0 >> 4;
    int lbase = (n0 & 15) + ((kp >= 8) ? 16 : 0);
    int v     = kp & 7;
    uint32_t* slot = &Bls[buf][(tt * 32 + lbase) * BSLOT + v];
    slot[0 * BSLOT] = (q0.x & 0xffffu) | (q1.x << 16);
    slot[1 * BSLOT] = (q0.x >> 16)     | (q1.x & 0xffff0000u);
    slot[2 * BSLOT] = (q0.y & 0xffffu) | (q1.y << 16);
    slot[3 * BSLOT] = (q0.y >> 16)     | (q1.y & 0xffff0000u);
  };

  // ---- prologue: stage tile k0=0 into buffer 0 ----
  stage_A(A + (size_t)m0 * lda, &Afs[0][0], lda, tid, wave);
  {
    uint2 q0[NTASKS], q1[NTASKS];
    #pragma unroll
    for (int j = 0; j < NTASKS; ++j) b_load(0, j, q0[j], q1[j]);
    #pragma unroll
    for (int j = 0; j < NTASKS; ++j) b_store(0, j, q0[j], q1[j]);
  }
  stage_A_fence(wave);
  __syncthreads();

  int cur = 0;
  for (int k0 = 0; k0 < K; k0 += GEMM_KB) {
    const bool have_next = (k0 + GEMM_KB) < K;
    uint2 q0[NTASKS], q1[NTASKS];
    if (have_next) {
      // issue next A tile DMA + next B global loads (overlap with compute)
      stage_A(A + (size_t)m0 * lda + (k0 + GEMM_KB), &Afs[cur ^ 1][0], lda,
              tid, wave);
      #pragma unroll
      for (int j = 0; j < NTASKS; ++j) b_load(k0 + GEMM_KB, j, q0[j], q1[j]);
    }

    // ---- A fragment: read 16 f32 from LDS, convert to bf16 in-register ----
    const float* ar = &Afs[cur][(wave * 16 + (lane & 15)) * APITCH
                               + ((lane < 16) ? 0 : 16)];
    float4 f0 = *reinterpret_cast<const float4*>(ar);
    float4 f1 = *reinterpret_cast<const float4*>(ar + 4);
    float4 f2 = *reinterpret_cast<const float4*>(ar + 8);
    float4 f3 = *reinterpret_cast<const float4*>(ar + 12);
    bf16x16 av;
    av[0]  = (__bf16)f0.x; av[1]  = (__bf16)f0.y;
    av[2]  = (__bf16)f0.z; av[3]  = (__bf16)f0.w;
    av[4]  = (__bf16)f1.x; av[5]  = (__bf16)f1.y;
    av[6]  = (__bf16)f1.z; av[7]  = (__bf16)f1.w;
    av[8]  = (__bf16)f2.x; av[9]  = (__bf16)f2.y;
    av[10] = (__bf16)f2.z; av[11] = (__bf16)f2.w;
    av[12] = (__bf16)f3.x; av[13] = (__bf16)f3.y;
    av[14] = (__bf16)f3.z; av[15] = (__bf16)f3.w;

    // ---- preload ALL B fragments, then issue WMMAs back-to-back ----
    // (accumulators are independent -> no D->A/B RAW hazard, no DS stalls
    //  in front of each WMMA)
    BFragU bfr[NT];
    #pragma unroll
    for (int t = 0; t < NT; ++t) {
      const uint32_t* bb = &Bls[cur][(t * 32 + lane) * BSLOT];
      bfr[t].u4[0] = *reinterpret_cast<const uint4*>(bb);
      bfr[t].u4[1] = *reinterpret_cast<const uint4*>(bb + 4);
    }
    #pragma unroll
    for (int t = 0; t < NT; ++t) {
      acc[t] = __builtin_amdgcn_wmma_f32_16x16x32_bf16(
          false, av, false, bfr[t].v, (short)0, acc[t], false, false);
    }

    if (have_next) {
      #pragma unroll
      for (int j = 0; j < NTASKS; ++j) b_store(cur ^ 1, j, q0[j], q1[j]);
    }
    stage_A_fence(wave);          // DMA for next buffer complete
    __syncthreads();              // B stores visible; cur reads all done
    cur ^= 1;
  }

  // ---- epilogue: C layout -> VGPR r: lanes0-15 (M=r,N=l), lanes16-31 (M=8+r)
  const int mbase = m0 + wave * 16 + ((lane < 16) ? 0 : 8);
  #pragma unroll
  for (int t = 0; t < NT; ++t) {
    const int n = t * 16 + (lane & 15);
    #pragma unroll
    for (int r = 0; r < 8; ++r) {
      const int m = mbase + r;
      float val = acc[t][r];
      if (rowscale) val *= rowscale[m];
      if (base)     val = 2.0f * val - base[(size_t)m * ldbase + n];
      C[(size_t)m * ldc + n] = val;
      if (Cbf) Cbf[(size_t)m * ldcb + n] = (__bf16)val;
    }
  }
}

// -------------- WMMA GEMM with transposed A (B1^T case only) ---------------
// C[M x N](f32) = A^T * B where A is f32 [K x M'] row-major (lda = M').
template<int NT>
__global__ __launch_bounds__(128)
void gemm_tA_wmma(const float* __restrict__ A, int lda,
                  const __bf16* __restrict__ B, int ldb, int K,
                  float* __restrict__ C, int ldc,
                  __bf16* __restrict__ Cbf, int ldcb,
                  const float* __restrict__ rowscale)
{
  constexpr int N     = NT * 16;
  constexpr int ASTR  = 40;
  constexpr int BSLOT = 12;
  __shared__ __align__(16) uint16_t Als[GEMM_MBLK * ASTR];
  __shared__ __align__(16) uint32_t Bls[NT * 32 * BSLOT];

  const int tid  = threadIdx.x;
  const int wave = tid >> 5;
  const int lane = tid & 31;
  const int m0   = blockIdx.x * GEMM_MBLK;

  f32x8 acc[NT];
  #pragma unroll
  for (int t = 0; t < NT; ++t) {
    #pragma unroll
    for (int j = 0; j < 8; ++j) acc[t][j] = 0.0f;
  }

  for (int k0 = 0; k0 < K; k0 += GEMM_KB) {
    #pragma unroll
    for (int r = 0; r < 4; ++r) {
      int idx = tid + r * 128;             // 512 = 32 k-rows x 16 groups of 4
      int k   = idx >> 4;
      int i0  = (idx & 15) << 2;
      const float* src = A + (size_t)(k0 + k) * lda + m0 + i0;
      float4 v = *reinterpret_cast<const float4*>(src);
      Als[(i0 + 0) * ASTR + k] = __builtin_bit_cast(uint16_t, (__bf16)v.x);
      Als[(i0 + 1) * ASTR + k] = __builtin_bit_cast(uint16_t, (__bf16)v.y);
      Als[(i0 + 2) * ASTR + k] = __builtin_bit_cast(uint16_t, (__bf16)v.z);
      Als[(i0 + 3) * ASTR + k] = __builtin_bit_cast(uint16_t, (__bf16)v.w);
      if (k0 + GEMM_KB < K) __builtin_prefetch(src + (size_t)GEMM_KB * lda, 0, 1);
    }
    {
      constexpr int NTASK = 16 * (N / 4);
      for (int t4 = tid; t4 < NTASK; t4 += 128) {
        int kp = t4 / (N / 4);
        int n0 = (t4 % (N / 4)) << 2;
        const __bf16* b0 = B + (size_t)(k0 + 2 * kp) * ldb + n0;
        uint2 q0 = *reinterpret_cast<const uint2*>(b0);
        uint2 q1 = *reinterpret_cast<const uint2*>(b0 + ldb);
        int tt    = n0 >> 4;
        int lbase = (n0 & 15) + ((kp >= 8) ? 16 : 0);
        int v     = kp & 7;
        uint32_t* slot = &Bls[(tt * 32 + lbase) * BSLOT + v];
        slot[0 * BSLOT] = (q0.x & 0xffffu) | (q1.x << 16);
        slot[1 * BSLOT] = (q0.x >> 16)     | (q1.x & 0xffff0000u);
        slot[2 * BSLOT] = (q0.y & 0xffffu) | (q1.y << 16);
        slot[3 * BSLOT] = (q0.y >> 16)     | (q1.y & 0xffff0000u);
      }
    }
    __syncthreads();

    const uint16_t* ab = &Als[(wave * 16 + (lane & 15)) * ASTR
                              + ((lane < 16) ? 0 : 16)];
    AFragU af;
    af.u4[0] = *reinterpret_cast<const uint4*>(ab);
    af.u4[1] = *reinterpret_cast<const uint4*>(ab + 8);

    BFragU bfr[NT];
    #pragma unroll
    for (int t = 0; t < NT; ++t) {
      const uint32_t* bb = &Bls[(t * 32 + lane) * BSLOT];
      bfr[t].u4[0] = *reinterpret_cast<const uint4*>(bb);
      bfr[t].u4[1] = *reinterpret_cast<const uint4*>(bb + 4);
    }
    #pragma unroll
    for (int t = 0; t < NT; ++t) {
      acc[t] = __builtin_amdgcn_wmma_f32_16x16x32_bf16(
          false, af.v, false, bfr[t].v, (short)0, acc[t], false, false);
    }
    __syncthreads();
  }

  const int mbase = m0 + wave * 16 + ((lane < 16) ? 0 : 8);
  #pragma unroll
  for (int t = 0; t < NT; ++t) {
    const int n = t * 16 + (lane & 15);
    #pragma unroll
    for (int r = 0; r < 8; ++r) {
      const int m = mbase + r;
      float val = acc[t][r] * rowscale[m];
      C[(size_t)m * ldc + n] = val;
      if (Cbf) Cbf[(size_t)m * ldcb + n] = (__bf16)val;
    }
  }
}

// ---- small helper kernels -------------------------------------------------

__global__ __launch_bounds__(256)
void prep_kernel(const float* __restrict__ D10, const float* __restrict__ D20,
                 float* __restrict__ d10inv, float* __restrict__ d20) {
  int i = blockIdx.x * 256 + threadIdx.x;          // grid covers 8192
  if (i < 2048) d10inv[i] = 1.0f / D10[(size_t)i * 2049];
  d20[i] = D20[(size_t)i * 8193];
}

__global__ __launch_bounds__(256)
void build0_kernel(const float* __restrict__ x0, const float* __restrict__ d10inv,
                   float* __restrict__ Base0, __bf16* __restrict__ RHS0,
                   __bf16* __restrict__ U0bf) {
  int idx = blockIdx.x * 256 + threadIdx.x;        // 2048*64
  int n = idx >> 6, f = idx & 63;
  float v = x0[idx];
  Base0[n * 128 + f] = v;
  RHS0[n * 128 + f]  = (__bf16)v;
  U0bf[idx]          = (__bf16)(d10inv[n] * v);
}

__global__ __launch_bounds__(256)
void build1_kernel(const float* __restrict__ x1,
                   float* __restrict__ Base1, __bf16* __restrict__ RHS1) {
  int idx = blockIdx.x * 256 + threadIdx.x;        // 8192*64
  int n = idx >> 6, f = idx & 63;
  float v = x1[idx];
  Base1[n * 128 + 64 + f] = v;
  RHS1[n * 128 + 64 + f]  = (__bf16)v;
}

// y[n,o] = relu( sum_i sum_c Base[n,64c+i]*W[i,o,3c] + T1*W[..,3c+1] + T2*W[..,3c+2] )
__global__ __launch_bounds__(256)
void final_kernel(const float* __restrict__ Base, const float* __restrict__ T1,
                  const float* __restrict__ T2, const float* __restrict__ W,
                  float* __restrict__ y, int Nrows) {
  int n = blockIdx.x * 4 + (threadIdx.x >> 6);
  int o = threadIdx.x & 63;
  if (n >= Nrows) return;
  const float* b  = Base + (size_t)n * 128;
  const float* t1 = T1   + (size_t)n * 128;
  const float* t2 = T2   + (size_t)n * 128;
  float s = 0.0f;
  #pragma unroll 4
  for (int i = 0; i < 64; ++i) {
    const float* w = W + ((size_t)i * 64 + o) * 6;
    s += b[i]      * w[0] + t1[i]      * w[1] + t2[i]      * w[2];
    s += b[64 + i] * w[3] + t1[64 + i] * w[4] + t2[64 + i] * w[5];
  }
  y[(size_t)n * 64 + o] = fmaxf(s, 0.0f);
}

// ---- host-side orchestration ---------------------------------------------

extern "C" void kernel_launch(void* const* d_in, const int* in_sizes, int n_in,
                              void* d_out, int out_size, void* d_ws, size_t ws_size,
                              hipStream_t stream) {
  (void)in_sizes; (void)n_in; (void)out_size; (void)ws_size;
  const float* x0  = (const float*)d_in[0];   // [2048,64]
  const float* x1  = (const float*)d_in[1];   // [8192,64]
  const float* L0  = (const float*)d_in[2];   // [2048,2048]
  const float* L1l = (const float*)d_in[3];   // [8192,8192]
  const float* B1  = (const float*)d_in[4];   // [2048,8192]
  const float* D10 = (const float*)d_in[5];   // [2048,2048] diag
  const float* D20 = (const float*)d_in[6];   // [8192,8192] diag
  const float* W0  = (const float*)d_in[7];   // [64,64,6]
  const float* W1  = (const float*)d_in[8];   // [64,64,6]
  float* out = (float*)d_out;

  char* ws = (char*)d_ws;
  size_t cur = 0;
  auto alloc = [&](size_t bytes) -> void* {
    void* p = ws + cur;
    cur = (cur + bytes + 255) & ~(size_t)255;
    return p;
  };
  float*  d10inv = (float*) alloc(2048 * 4);
  float*  d20v   = (float*) alloc(8192 * 4);
  __bf16* U0bf   = (__bf16*)alloc(2048 * 64 * 2);
  float*  Base0  = (float*) alloc((size_t)2048 * 128 * 4);  // [x0 | x0p]
  __bf16* RHS0   = (__bf16*)alloc((size_t)2048 * 128 * 2);
  float*  T1_0   = (float*) alloc((size_t)2048 * 128 * 4);
  __bf16* T1_0bf = (__bf16*)alloc((size_t)2048 * 128 * 2);
  float*  T2_0   = (float*) alloc((size_t)2048 * 128 * 4);
  float*  Base1  = (float*) alloc((size_t)8192 * 128 * 4);  // [x1n | x1]
  __bf16* RHS1   = (__bf16*)alloc((size_t)8192 * 128 * 2);
  float*  T1_1   = (float*) alloc((size_t)8192 * 128 * 4);
  __bf16* T1_1bf = (__bf16*)alloc((size_t)8192 * 128 * 2);
  float*  T2_1   = (float*) alloc((size_t)8192 * 128 * 4);

  prep_kernel  <<<32,   256, 0, stream>>>(D10, D20, d10inv, d20v);
  build0_kernel<<<512,  256, 0, stream>>>(x0, d10inv, Base0, RHS0, U0bf);
  build1_kernel<<<2048, 256, 0, stream>>>(x1, Base1, RHS1);

  // x0p = d10inv ⊙ (B1 @ x1)            -> Base0[:,64:], RHS0[:,64:]
  gemm_tdm_wmma<4><<<2048 / GEMM_MBLK, 128, 0, stream>>>(
      B1, 8192, RHS1 + 64, 128, 8192,
      Base0 + 64, 128, RHS0 + 64, 128, d10inv, nullptr, 0);

  // x1n = d20 ⊙ (B1^T @ (d10inv ⊙ x0))  -> Base1[:,0:64], RHS1[:,0:64]
  gemm_tA_wmma<4><<<8192 / GEMM_MBLK, 128, 0, stream>>>(
      B1, 8192, U0bf, 64, 2048,
      Base1, 128, RHS1, 128, d20v);

  // T1_0 = L0 @ [x0 | x0p]
  gemm_tdm_wmma<8><<<2048 / GEMM_MBLK, 128, 0, stream>>>(
      L0, 2048, RHS0, 128, 2048,
      T1_0, 128, T1_0bf, 128, nullptr, nullptr, 0);

  // T2_0 = 2*(L0 @ T1_0) - [x0 | x0p]
  gemm_tdm_wmma<8><<<2048 / GEMM_MBLK, 128, 0, stream>>>(
      L0, 2048, T1_0bf, 128, 2048,
      T2_0, 128, nullptr, 0, nullptr, Base0, 128);

  // T1_1 = L1l @ [x1n | x1]
  gemm_tdm_wmma<8><<<8192 / GEMM_MBLK, 128, 0, stream>>>(
      L1l, 8192, RHS1, 128, 8192,
      T1_1, 128, T1_1bf, 128, nullptr, nullptr, 0);

  // T2_1 = 2*(L1l @ T1_1) - [x1n | x1]
  gemm_tdm_wmma<8><<<8192 / GEMM_MBLK, 128, 0, stream>>>(
      L1l, 8192, T1_1bf, 128, 8192,
      T2_1, 128, nullptr, 0, nullptr, Base1, 128);

  final_kernel<<<2048 / 4, 256, 0, stream>>>(Base0, T1_0, T2_0, W0, out, 2048);
  final_kernel<<<8192 / 4, 256, 0, stream>>>(Base1, T1_1, T2_1, W1,
                                             out + 2048 * 64, 8192);
}
